// CAAM_25494925869251
// MI455X (gfx1250) — matrix operand order repaired and checked by reading, not compiled, not run on hardware
//
#include <hip/hip_runtime.h>
#include <math.h>

#define WB   8
#define WC   512
#define WW   128
#define WK   19
#define WHW  8192
#define WN   8
#define WCI  256

typedef __attribute__((ext_vector_type(16))) _Float16 v16h;
typedef __attribute__((ext_vector_type(8)))  float    v8f;

__device__ __forceinline__ v8f wmma_f16(v16h a, v16h b, v8f c) {
  // 8 args: (neg_a, A, neg_b, B, c_mod, C, reuse_a, reuse_b)
  return __builtin_amdgcn_wmma_f32_16x16x32_f16(false, a, false, b, (short)0, c, false, false);
}

// Async global -> LDS copy, 16 bytes per lane (tracked by ASYNCcnt).
// Generic LDS pointer low 32 bits == LDS byte offset (aperture layout, ISA 10.2).
__device__ __forceinline__ void async_g2l_b128(void* lds, const void* g) {
  asm volatile("global_load_async_to_lds_b128 %0, %1, off"
               :: "v"((unsigned)(uintptr_t)lds), "v"(g) : "memory");
}
__device__ __forceinline__ void wait_asynccnt0() {
  asm volatile("s_wait_asynccnt 0" ::: "memory");
}

// A fragment, 16x32 f16, from LDS f32 tile (row-major, leading dim ld).
// lane l: M = l&15, half h = l>>4; VGPR j holds K = 16*(j/4) + 8*h + 2*(j%4), +1
__device__ __forceinline__ v16h frag_a16x32(const float* As, int ld, int lane) {
  int m = lane & 15, h = lane >> 4;
  v16h a;
#pragma unroll
  for (int j = 0; j < 8; ++j) {
    int k = ((j >> 2) << 4) + (h << 3) + ((j & 3) << 1);
    a[2 * j]     = (_Float16)As[m * ld + k];
    a[2 * j + 1] = (_Float16)As[m * ld + k + 1];
  }
  return a;
}

// B fragment, 32x16 f16, from LDS f32 tile (K-major rows, leading dim ld).
// lane l: N = l&15, half h = l>>4; VGPR j holds K = 16*(j/4) + 8*h + 2*(j%4), +1
__device__ __forceinline__ v16h frag_b32x16(const float* Bs, int ld, int lane) {
  int n = lane & 15, h = lane >> 4;
  v16h b;
#pragma unroll
  for (int j = 0; j < 8; ++j) {
    int k = ((j >> 2) << 4) + (h << 3) + ((j & 3) << 1);
    b[2 * j]     = (_Float16)Bs[k * ld + n];
    b[2 * j + 1] = (_Float16)Bs[(k + 1) * ld + n];
  }
  return b;
}

// ---------------- K1: cam[b,k,hw] = Wcam(19x512) @ x[b](512x8192) + bias ----------------
__global__ void k1_cam(const float* __restrict__ x, const float* __restrict__ wcam,
                       const float* __restrict__ bcam, float* __restrict__ cam) {
  int lane = threadIdx.x;
  int blk = blockIdx.x;
  int b = blk / (WHW / 16);
  int p0 = (blk % (WHW / 16)) * 16;
  __shared__ float As[1024];
  __shared__ float Bs[512];
  v8f z = {0.f,0.f,0.f,0.f,0.f,0.f,0.f,0.f};
  v8f acc0 = z, acc1 = z;
  for (int c0 = 0; c0 < WC; c0 += 32) {
    for (int i = lane; i < 1024; i += 32) {
      int t = i >> 9, r = (i >> 5) & 15, k = i & 31;
      int cls = t * 16 + r;
      As[i] = (cls < WK) ? wcam[cls * WC + c0 + k] : 0.f;
    }
    {
      // B tile: one row (channel) per lane, 4 x b128 async copies of 16 pixels
      const float* src = &x[((size_t)b * WC + c0 + lane) * WHW + p0];
#pragma unroll
      for (int s = 0; s < 4; ++s)
        async_g2l_b128(&Bs[lane * 16 + s * 4], src + s * 4);
    }
    wait_asynccnt0();
    __syncthreads();
    v16h bf = frag_b32x16(Bs, 16, lane);
    acc0 = wmma_f16(frag_a16x32(As, 32, lane), bf, acc0);
    acc1 = wmma_f16(frag_a16x32(As + 512, 32, lane), bf, acc1);
    __syncthreads();
  }
  int n = lane & 15, h = lane >> 4;
#pragma unroll
  for (int r = 0; r < 8; ++r) {
    int m = r + 8 * h;
    cam[((size_t)b * WK + m) * WHW + p0 + n] = acc0[r] + bcam[m];
    int m2 = m + 16;
    if (m2 < WK) cam[((size_t)b * WK + m2) * WHW + p0 + n] = acc1[r] + bcam[m2];
  }
}

// ---------------- K2: per (b,n,k): bin mean->sigmoid conf, max, 1/sum(exp) --------------
__global__ void k2_binstats(const float* __restrict__ cam, float* __restrict__ conf,
                            float* __restrict__ maxv, float* __restrict__ invs) {
  int tid = threadIdx.x;
  int blk = blockIdx.x;               // b*WN*WK + n*WK + k
  int b = blk / (WN * WK);
  int n = (blk / WK) % WN;
  int k = blk % WK;
  int bh = n >> 2, bw = n & 3;
  size_t base = ((size_t)b * WK + k) * WHW + (size_t)(bh * 32) * WW + bw * 32;
  float v[4];
#pragma unroll
  for (int j = 0; j < 4; ++j) {
    int p = tid + 256 * j;
    v[j] = cam[base + (size_t)(p >> 5) * WW + (p & 31)];
  }
  __shared__ float red[256];
  red[tid] = v[0] + v[1] + v[2] + v[3];
  __syncthreads();
  for (int o = 128; o > 0; o >>= 1) { if (tid < o) red[tid] += red[tid + o]; __syncthreads(); }
  float tot = red[0]; __syncthreads();
  red[tid] = fmaxf(fmaxf(v[0], v[1]), fmaxf(v[2], v[3]));
  __syncthreads();
  for (int o = 128; o > 0; o >>= 1) { if (tid < o) red[tid] = fmaxf(red[tid], red[tid + o]); __syncthreads(); }
  float mx = red[0]; __syncthreads();
  red[tid] = expf(v[0]-mx) + expf(v[1]-mx) + expf(v[2]-mx) + expf(v[3]-mx);
  __syncthreads();
  for (int o = 128; o > 0; o >>= 1) { if (tid < o) red[tid] += red[tid + o]; __syncthreads(); }
  if (tid == 0) {
    conf[blk] = 1.f / (1.f + expf(-tot * (1.f / 1024.f)));
    maxv[blk] = mx;
    invs[blk] = 1.f / red[0];
  }
}

// ---------------- K3: local[b,n,k,c] = conf*invs * sum_p exp(cam-max) * x ---------------
__global__ void k3_local(const float* __restrict__ cam, const float* __restrict__ x,
                         const float* __restrict__ conf, const float* __restrict__ maxv,
                         const float* __restrict__ invs, float* __restrict__ local) {
  int lane = threadIdx.x;
  int blk = blockIdx.x;               // b*WN*32 + n*32 + ct
  int b = blk / (WN * 32);
  int n = (blk / 32) % WN;
  int c0 = (blk & 31) * 16;
  int bh = n >> 2, bw = n & 3;
  size_t pixbase = (size_t)(bh * 32) * WW + bw * 32;
  __shared__ float As[1024];
  __shared__ float Bs[512];
  __shared__ float mxv[32];
  mxv[lane] = (lane < WK) ? maxv[((size_t)b * WN + n) * WK + lane] : 0.f;
  __syncthreads();
  v8f z = {0.f,0.f,0.f,0.f,0.f,0.f,0.f,0.f};
  v8f acc0 = z, acc1 = z;
  for (int it = 0; it < 32; ++it) {   // 32 pixels (one bin row) per K-step
    size_t rowoff = pixbase + (size_t)it * WW;
    for (int i = lane; i < 1024; i += 32) {
      int t = i >> 9, r = (i >> 5) & 15, pk = i & 31;
      int cls = t * 16 + r;
      As[i] = (cls < WK) ? expf(cam[((size_t)b * WK + cls) * WHW + rowoff + pk] - mxv[cls]) : 0.f;
    }
    for (int i = lane; i < 512; i += 32) {
      int p = i >> 4, c = i & 15;
      Bs[i] = x[((size_t)b * WC + c0 + c) * WHW + rowoff + p];
    }
    __syncthreads();
    v16h bf = frag_b32x16(Bs, 16, lane);
    acc0 = wmma_f16(frag_a16x32(As, 32, lane), bf, acc0);
    acc1 = wmma_f16(frag_a16x32(As + 512, 32, lane), bf, acc1);
    __syncthreads();
  }
  int col = lane & 15, h = lane >> 4;
#pragma unroll
  for (int r = 0; r < 8; ++r) {
    int m = r + 8 * h;
    float sc = conf[((size_t)b*WN+n)*WK + m] * invs[((size_t)b*WN+n)*WK + m];
    local[(((size_t)b*WN+n)*WK + m)*WC + c0 + col] = acc0[r] * sc;
    int m2 = m + 16;
    if (m2 < WK) {
      float sc2 = conf[((size_t)b*WN+n)*WK + m2] * invs[((size_t)b*WN+n)*WK + m2];
      local[(((size_t)b*WN+n)*WK + m2)*WC + c0 + col] = acc1[r] * sc2;
    }
  }
}

// ---------------- K4: GCN bin-mix + residual PReLU (tiny, VALU) -------------------------
__global__ void k4_gcn1(const float* __restrict__ local, const float* __restrict__ w1,
                        const float* __restrict__ ga, float* __restrict__ tbuf) {
  int idx = blockIdx.x * blockDim.x + threadIdx.x;   // B*WK*WC
  if (idx >= WB * WK * WC) return;
  int b = idx / (WK * WC);
  int k = (idx / WC) % WK;
  int c = idx % WC;
  float lm[WN];
#pragma unroll
  for (int m = 0; m < WN; ++m)
    lm[m] = local[(((size_t)b * WN + m) * WK + k) * WC + c];
#pragma unroll
  for (int n = 0; n < WN; ++n) {
    float s = lm[n];
#pragma unroll
    for (int m = 0; m < WN; ++m) s += w1[n * WN + m] * lm[m];
    float a = ga[n];
    s = fmaxf(s, 0.f) + a * fminf(s, 0.f);
    tbuf[(((size_t)b * WN + n) * WK + k) * WC + c] = s;
  }
}

// ---------------- Generic batched WMMA GEMM: C(19 x ntiles*16) = A(19 x K) @ B ----------
// bT=1: B[r,col] = Bm[col*ldb + r]   (weight stored [col][r])
// bT=0: B[r,col] = Bm[r*ldb + col]
__global__ void k_gemm19(const float* __restrict__ Abase, long bsA, int lda,
                         const float* __restrict__ Bm, int ldb, int bT,
                         const float* __restrict__ bias,
                         float* __restrict__ Cbase, long bsC, int ldc,
                         int Kdim, int ntiles) {
  int lane = threadIdx.x;
  int blk = blockIdx.x;
  int batch = blk / ntiles;
  int n0 = (blk % ntiles) * 16;
  const float* A = Abase + (size_t)batch * bsA;
  float* C = Cbase + (size_t)batch * bsC;
  __shared__ float As[1024];
  __shared__ float Bs[512];
  v8f z = {0.f,0.f,0.f,0.f,0.f,0.f,0.f,0.f};
  v8f acc0 = z, acc1 = z;
  for (int k0 = 0; k0 < Kdim; k0 += 32) {
    for (int i = lane; i < 1024; i += 32) {
      int t = i >> 9, r = (i >> 5) & 15, kk = i & 31;
      int cls = t * 16 + r;
      As[i] = (cls < WK) ? A[(size_t)cls * lda + k0 + kk] : 0.f;
    }
    for (int i = lane; i < 512; i += 32) {
      int r = i >> 4, nn = i & 15;
      Bs[i] = bT ? Bm[(size_t)(n0 + nn) * ldb + k0 + r]
                 : Bm[(size_t)(k0 + r) * ldb + n0 + nn];
    }
    __syncthreads();
    v16h bf = frag_b32x16(Bs, 16, lane);
    acc0 = wmma_f16(frag_a16x32(As, 32, lane), bf, acc0);
    acc1 = wmma_f16(frag_a16x32(As + 512, 32, lane), bf, acc1);
    __syncthreads();
  }
  int col = lane & 15, h = lane >> 4;
  float bv = bias ? bias[n0 + col] : 0.f;
#pragma unroll
  for (int r = 0; r < 8; ++r) {
    int m = r + 8 * h;
    C[(size_t)m * ldc + n0 + col] = acc0[r] + bv;
    int m2 = m + 16;
    if (m2 < WK) C[(size_t)m2 * ldc + n0 + col] = acc1[r] + bv;
  }
}

// ---------------- K6: fuse bins -> glob, PReLU (tiny) -----------------------------------
__global__ void k6_fuse(const float* __restrict__ local2, const float* __restrict__ fw,
                        const float* __restrict__ fb, const float* __restrict__ ra,
                        float* __restrict__ glob) {
  int idx = blockIdx.x * blockDim.x + threadIdx.x;   // B*WK*WC
  if (idx >= WB * WK * WC) return;
  int b = idx / (WK * WC);
  int k = (idx / WC) % WK;
  int c = idx % WC;
  float s = fb[0];
#pragma unroll
  for (int n = 0; n < WN; ++n)
    s += fw[n] * local2[(((size_t)b * WN + n) * WK + k) * WC + c];
  float a = ra[0];
  s = fmaxf(s, 0.f) + a * fminf(s, 0.f);
  glob[((size_t)b * WK + k) * WC + c] = s;
}

// ---------------- K7c: logit bias d[b,n,k] = key . q_b ----------------------------------
__global__ void k7c_dbias(const float* __restrict__ key, const float* __restrict__ qb,
                          float* __restrict__ dbias) {
  int idx = blockIdx.x * blockDim.x + threadIdx.x;   // B*WN*WK
  if (idx >= WB * WN * WK) return;
  const float* kr = key + (size_t)idx * WCI;
  float s = 0.f;
  for (int i = 0; i < WCI; ++i) s += kr[i] * qb[i];
  dbias[idx] = s;
}

// ---------------- K9: fused attention logits->softmax->out->out_conv, y + BN stats ------
__global__ void k9_attn_out(const float* __restrict__ x, const float* __restrict__ kk,
                            const float* __restrict__ dbias, const float* __restrict__ val,
                            const float* __restrict__ outw, float* __restrict__ yout,
                            float* __restrict__ gsum, float* __restrict__ gsq) {
  int lane = threadIdx.x;
  int blk = blockIdx.x;               // b*512 + pixel tile
  int b = blk >> 9;
  int p0 = (blk & 511) * 16;
  int hh = p0 >> 7, w0 = p0 & 127;
  int n = ((hh >> 5) << 2) + (w0 >> 5);
  __shared__ float As[1024];
  __shared__ float Bs[512];
  __shared__ float Aff[512];          // 32 (k, padded) x 16 (pixels)
  __shared__ float Out[4096];         // 256 (i) x 16 (pixels)
  __shared__ float Csum[512];
  __shared__ float Csq[512];
  for (int i = lane; i < 512; i += 32) { Csum[i] = 0.f; Csq[i] = 0.f; Aff[i] = 0.f; }
  const float* kkb = kk + ((size_t)b * WN + n) * WK * WC;
  v8f z = {0.f,0.f,0.f,0.f,0.f,0.f,0.f,0.f};
  v8f acc0 = z, acc1 = z;
  // Phase 1: logits[k, p] = sum_c kk[k,c] * x[c,p]
  for (int c0 = 0; c0 < WC; c0 += 32) {
    for (int i = lane; i < 1024; i += 32) {
      int t = i >> 9, r = (i >> 5) & 15, ci = i & 31;
      int cls = t * 16 + r;
      As[i] = (cls < WK) ? kkb[(size_t)cls * WC + c0 + ci] : 0.f;
    }
    {
      const float* src = &x[((size_t)b * WC + c0 + lane) * WHW + p0];
#pragma unroll
      for (int s = 0; s < 4; ++s)
        async_g2l_b128(&Bs[lane * 16 + s * 4], src + s * 4);
    }
    if (c0 + 32 < WC)  // hint next K-step's x rows into the caches
      __builtin_prefetch(&x[((size_t)b * WC + c0 + 32 + lane) * WHW + p0], 0, 0);
    wait_asynccnt0();
    __syncthreads();
    v16h bf = frag_b32x16(Bs, 16, lane);
    acc0 = wmma_f16(frag_a16x32(As, 32, lane), bf, acc0);
    acc1 = wmma_f16(frag_a16x32(As + 512, 32, lane), bf, acc1);
    __syncthreads();
  }
  {
    int col = lane & 15, h = lane >> 4;
    const float* db = dbias + ((size_t)b * WN + n) * WK;
#pragma unroll
    for (int r = 0; r < 8; ++r) {
      int m = r + 8 * h;
      Aff[m * 16 + col] = acc0[r] + db[m];
      int m2 = m + 16;
      if (m2 < WK) Aff[m2 * 16 + col] = acc1[r] + db[m2];
    }
  }
  __syncthreads();
  // softmax over k (one pixel column per lane); padded rows stay 0
  if (lane < 16) {
    float mx = -3.0e38f;
    for (int k = 0; k < WK; ++k) mx = fmaxf(mx, Aff[k * 16 + lane]);
    float s = 0.f;
    for (int k = 0; k < WK; ++k) s += expf(Aff[k * 16 + lane] - mx);
    float inv = 1.f / s;
    for (int k = 0; k < WK; ++k) Aff[k * 16 + lane] = expf(Aff[k * 16 + lane] - mx) * inv;
  }
  __syncthreads();
  // Phase 2: out[i,p] = sum_k val[k,i] * aff[k,p]
  v16h affFrag = frag_b32x16(Aff, 16, lane);
  const float* valb = val + (size_t)b * WK * WCI;
  for (int it = 0; it < 16; ++it) {
    for (int i = lane; i < 512; i += 32) {
      int m = i >> 5, kidx = i & 31;
      As[i] = (kidx < WK) ? valb[(size_t)kidx * WCI + it * 16 + m] : 0.f;
    }
    __syncthreads();
    v8f o = wmma_f16(frag_a16x32(As, 32, lane), affFrag, z);
    int col = lane & 15, h = lane >> 4;
#pragma unroll
    for (int r = 0; r < 8; ++r)
      Out[(it * 16 + r + 8 * h) * 16 + col] = o[r];
    __syncthreads();
  }
  // Phase 3: y[c,p] = sum_i out_w[c,i] * out[i,p]; write raw y, accumulate stats
  for (int ct = 0; ct < 32; ++ct) {
    v8f acc = z;
    for (int i0 = 0; i0 < WCI; i0 += 32) {
      for (int i = lane; i < 512; i += 32) {
        int m = i >> 5, ii = i & 31;
        As[i] = outw[(size_t)(ct * 16 + m) * WCI + i0 + ii];
      }
      __syncthreads();
      acc = wmma_f16(frag_a16x32(As, 32, lane), frag_b32x16(&Out[i0 * 16], 16, lane), acc);
      __syncthreads();
    }
    int col = lane & 15, h = lane >> 4;
#pragma unroll
    for (int r = 0; r < 8; ++r) {
      int m = ct * 16 + r + 8 * h;
      float v = acc[r];
      yout[((size_t)b * WC + m) * WHW + p0 + col] = v;
      atomicAdd(&Csum[m], v);
      atomicAdd(&Csq[m], v * v);
    }
  }
  __syncthreads();
  for (int i = lane; i < 512; i += 32) {
    atomicAdd(&gsum[i], Csum[i]);
    atomicAdd(&gsq[i], Csq[i]);
  }
}

// ---------------- K10: finalize BN affine --------------------------------------------
__global__ void k10_bnfin(const float* __restrict__ gsum, const float* __restrict__ gsq,
                          const float* __restrict__ gamma, const float* __restrict__ beta,
                          float* __restrict__ bnA, float* __restrict__ bnB) {
  int c = threadIdx.x;
  float cnt = (float)(WB * WHW);
  float mu = gsum[c] / cnt;
  float var = gsq[c] / cnt - mu * mu;
  float a = gamma[c] * rsqrtf(var + 1e-5f);
  bnA[c] = a;
  bnB[c] = beta[c] - mu * a;
}

// ---------------- K11: out = x + PReLU(bn(y)) (y already in d_out) ----------------------
__global__ void k11_final(const float* __restrict__ x, const float* __restrict__ bnA,
                          const float* __restrict__ bnB, const float* __restrict__ oa,
                          float* __restrict__ out) {
  size_t idx = (size_t)blockIdx.x * blockDim.x + threadIdx.x;
  size_t tot4 = (size_t)WB * WC * WHW / 4;
  if (idx >= tot4) return;
  int c = (int)((idx >> 11) & (WC - 1));     // HW/4 = 2048 float4 per (b,c)
  float a = bnA[c], bsh = bnB[c], al = oa[c];
  const float4* x4 = (const float4*)x;
  float4* o4 = (float4*)out;
  float4 xv = x4[idx];
  float4 yv = o4[idx];
  float f;
  f = a * yv.x + bsh; yv.x = xv.x + fmaxf(f, 0.f) + al * fminf(f, 0.f);
  f = a * yv.y + bsh; yv.y = xv.y + fmaxf(f, 0.f) + al * fminf(f, 0.f);
  f = a * yv.z + bsh; yv.z = xv.z + fmaxf(f, 0.f) + al * fminf(f, 0.f);
  f = a * yv.w + bsh; yv.w = xv.w + fmaxf(f, 0.f) + al * fminf(f, 0.f);
  o4[idx] = yv;
}

extern "C" void kernel_launch(void* const* d_in, const int* in_sizes, int n_in,
                              void* d_out, int out_size, void* d_ws, size_t ws_size,
                              hipStream_t stream) {
  (void)in_sizes; (void)n_in; (void)out_size; (void)ws_size;
  const float* x     = (const float*)d_in[0];
  const float* wcam  = (const float*)d_in[1];
  const float* bcam  = (const float*)d_in[2];
  const float* gw1   = (const float*)d_in[3];
  const float* ga    = (const float*)d_in[4];
  const float* gw2   = (const float*)d_in[5];
  const float* fw    = (const float*)d_in[6];
  const float* fb    = (const float*)d_in[7];
  const float* ra    = (const float*)d_in[8];
  const float* qw    = (const float*)d_in[9];
  const float* qb    = (const float*)d_in[10];
  const float* kw    = (const float*)d_in[11];
  const float* kb    = (const float*)d_in[12];
  const float* vw    = (const float*)d_in[13];
  const float* vb    = (const float*)d_in[14];
  const float* ow    = (const float*)d_in[15];
  const float* gamma = (const float*)d_in[16];
  const float* beta  = (const float*)d_in[17];
  const float* oa    = (const float*)d_in[18];
  float* out = (float*)d_out;
  float* ws = (float*)d_ws;

  size_t off = 0;
  float* cam    = ws + off; off += (size_t)WB * WK * WHW;
  float* conf   = ws + off; off += WB * WN * WK;
  float* maxv   = ws + off; off += WB * WN * WK;
  float* invs   = ws + off; off += WB * WN * WK;
  float* local  = ws + off; off += (size_t)WB * WN * WK * WC;
  float* tbuf   = ws + off; off += (size_t)WB * WN * WK * WC;
  float* local2 = ws + off; off += (size_t)WB * WN * WK * WC;
  float* keyb   = ws + off; off += (size_t)WB * WN * WK * WCI;
  float* kkb    = ws + off; off += (size_t)WB * WN * WK * WC;
  float* dbias  = ws + off; off += WB * WN * WK;
  float* globb  = ws + off; off += (size_t)WB * WK * WC;
  float* valb   = ws + off; off += (size_t)WB * WK * WCI;
  float* gsum   = ws + off; off += 512;
  float* gsq    = ws + off; off += 512;
  float* bnA    = ws + off; off += 512;
  float* bnB    = ws + off; off += 512;

  k1_cam<<<WB * (WHW / 16), 32, 0, stream>>>(x, wcam, bcam, cam);
  k2_binstats<<<WB * WN * WK, 256, 0, stream>>>(cam, conf, maxv, invs);
  k3_local<<<WB * WN * 32, 32, 0, stream>>>(cam, x, conf, maxv, invs, local);
  {
    int tot = WB * WK * WC;
    k4_gcn1<<<(tot + 255) / 256, 256, 0, stream>>>(local, gw1, ga, tbuf);
  }
  // local2[b,n,k,d] = sum_c t[b,n,k,c] * w2[d,c]
  k_gemm19<<<WB * WN * 32, 32, 0, stream>>>(tbuf, (long)WK * WC, WC, gw2, WC, 1,
                                            nullptr, local2, (long)WK * WC, WC, WC, 32);
  {
    int tot = WB * WK * WC;
    k6_fuse<<<(tot + 255) / 256, 256, 0, stream>>>(local2, fw, fb, ra, globb);
  }
  // key[b,n,k,i] = sum_c local2 * k_w[i,c] + k_b
  k_gemm19<<<WB * WN * 16, 32, 0, stream>>>(local2, (long)WK * WC, WC, kw, WC, 1,
                                            kb, keyb, (long)WK * WCI, WCI, WC, 16);
  // kk[b,n,k,c] = sum_i key * q_w[i,c]
  k_gemm19<<<WB * WN * 32, 32, 0, stream>>>(keyb, (long)WK * WCI, WCI, qw, WC, 0,
                                            nullptr, kkb, (long)WK * WC, WC, WCI, 32);
  {
    int tot = WB * WN * WK;
    k7c_dbias<<<(tot + 255) / 256, 256, 0, stream>>>(keyb, qb, dbias);
  }
  // val[b,k,i] = sum_c glob * v_w[i,c] + v_b
  k_gemm19<<<WB * 16, 32, 0, stream>>>(globb, (long)WK * WC, WC, vw, WC, 1,
                                       vb, valb, (long)WK * WCI, WCI, WC, 16);
  hipMemsetAsync(gsum, 0, 2 * 512 * sizeof(float), stream);
  k9_attn_out<<<WB * (WHW / 16), 32, 0, stream>>>(x, kkb, dbias, valb, ow, out, gsum, gsq);
  k10_bnfin<<<1, 512, 0, stream>>>(gsum, gsq, gamma, beta, bnA, bnB);
  {
    size_t tot4 = (size_t)WB * WC * WHW / 4;
    k11_final<<<(unsigned)((tot4 + 255) / 256), 256, 0, stream>>>(x, bnA, bnB, oa, out);
  }
}
